// MHA_40089224741264
// MI455X (gfx1250) — matrix operand order repaired and checked by reading, not compiled, not run on hardware
//
#include <hip/hip_runtime.h>
#include <hip/hip_bf16.h>

// MHA: B=4, T=2048, D=1024, H=16, DH=64
// qkv_gemm (f32->f16 WMMA GEMM) -> attn (flash WMMA + async K/V tiles) -> out_gemm

typedef __attribute__((ext_vector_type(16))) _Float16 v16h;
typedef __attribute__((ext_vector_type(8)))  _Float16 v8h;
typedef __attribute__((ext_vector_type(8)))  float    v8f;
typedef __attribute__((ext_vector_type(4)))  float    v4f;

#define BATCH 4
#define SEQ   2048
#define DMODEL 1024
#define NHEAD 16
#define DHEAD 64
#define NROWS (BATCH * SEQ)            // 8192
#define ELEMS ((size_t)NROWS * DMODEL) // 8388608 per f16 buffer

static __device__ inline v8f vzero8() {
  v8f z;
#pragma unroll
  for (int i = 0; i < 8; ++i) z[i] = 0.0f;
  return z;
}

static __device__ inline v16h frag16(const _Float16* p0, const _Float16* p1) {
  v8h a = *(const v8h*)p0;
  v8h b = *(const v8h*)p1;
  v16h r;
#pragma unroll
  for (int i = 0; i < 8; ++i) { r[i] = a[i]; r[i + 8] = b[i]; }
  return r;
}

static __device__ inline v8f wmma16(v16h a, v16h b, v8f c) {
  return __builtin_amdgcn_wmma_f32_16x16x32_f16(false, a, false, b, (short)0, c,
                                                false, false);
}

static __device__ inline float redmax16(float v) {
#pragma unroll
  for (int m = 1; m < 16; m <<= 1) v = fmaxf(v, __shfl_xor(v, m, 16));
  return v;
}
static __device__ inline float redsum16(float v) {
#pragma unroll
  for (int m = 1; m < 16; m <<= 1) v += __shfl_xor(v, m, 16);
  return v;
}

// One per-lane async 16B global->LDS copy (ASYNCcnt-tracked).
static __device__ inline void async_b128(const _Float16* gsrc, _Float16* ldst) {
  uint32_t lds_off = (uint32_t)(uintptr_t)ldst;          // LDS aperture low bits
  uint64_t gaddr = (uint64_t)(uintptr_t)gsrc;
  asm volatile("global_load_async_to_lds_b128 %0, %1, off"
               :
               : "v"(lds_off), "v"(gaddr)
               : "memory");
}
static __device__ inline void wait_async0() {
  asm volatile("s_wait_asynccnt 0x0" ::: "memory");
}

// ---------------------------------------------------------------------------
// Kernel 1: fused QKV projection.  C[8192,1024] = X @ W^T (+bias), f16 out.
// which = blockIdx.z: 0->Q [B,H,T,DH], 1->K [B,H,T,DH], 2->V transposed [B,H,DH,T]
// ---------------------------------------------------------------------------
__global__ __launch_bounds__(256) void qkv_gemm(
    const float* __restrict__ q, const float* __restrict__ k,
    const float* __restrict__ v, const float* __restrict__ Wq,
    const float* __restrict__ Wk, const float* __restrict__ Wv,
    const float* __restrict__ bq, const float* __restrict__ bk,
    const float* __restrict__ bv, _Float16* __restrict__ Qh,
    _Float16* __restrict__ Kh, _Float16* __restrict__ Vt) {
  const int which = blockIdx.z;
  const float* X = (which == 0) ? q : (which == 1) ? k : v;
  const float* W = (which == 0) ? Wq : (which == 1) ? Wk : Wv;
  const float* bias = (which == 0) ? bq : (which == 1) ? bk : bv;

  __shared__ _Float16 Xs[128 * 40];
  __shared__ _Float16 Ws[128 * 40];

  const int tid = threadIdx.x;
  const int lane = tid & 31;
  const int wave = tid >> 5;
  const int l = lane & 15;
  const int hh = lane >> 4;
  const int m0 = blockIdx.x * 128;
  const int n0 = blockIdx.y * 128;
  const int wm = (wave & 1) * 64;
  const int wn = (wave >> 1) * 32;

  v8f acc[4][2];
#pragma unroll
  for (int mi = 0; mi < 4; ++mi)
#pragma unroll
    for (int ni = 0; ni < 2; ++ni) acc[mi][ni] = vzero8();

  for (int k0 = 0; k0 < DMODEL; k0 += 32) {
#pragma unroll
    for (int i = 0; i < 4; ++i) {
      int idx = tid + i * 256;
      int row = idx >> 3;
      int c4 = (idx & 7) * 4;
      v4f x4 = *(const v4f*)(X + (size_t)(m0 + row) * DMODEL + k0 + c4);
      v4f w4 = *(const v4f*)(W + (size_t)(n0 + row) * DMODEL + k0 + c4);
      _Float16* dx = Xs + row * 40 + c4;
      _Float16* dw = Ws + row * 40 + c4;
#pragma unroll
      for (int j = 0; j < 4; ++j) { dx[j] = (_Float16)x4[j]; dw[j] = (_Float16)w4[j]; }
    }
    __syncthreads();

    v16h af[4], bf[2];
#pragma unroll
    for (int mi = 0; mi < 4; ++mi) {
      const _Float16* ar = Xs + (wm + mi * 16 + l) * 40;
      af[mi] = frag16(ar + hh * 8, ar + 16 + hh * 8);
    }
#pragma unroll
    for (int ni = 0; ni < 2; ++ni) {
      const _Float16* br = Ws + (wn + ni * 16 + l) * 40;
      bf[ni] = frag16(br + hh * 8, br + 16 + hh * 8);
    }
#pragma unroll
    for (int mi = 0; mi < 4; ++mi)
#pragma unroll
      for (int ni = 0; ni < 2; ++ni)
        acc[mi][ni] = wmma16(af[mi], bf[ni], acc[mi][ni]);
    __syncthreads();
  }

#pragma unroll
  for (int mi = 0; mi < 4; ++mi)
#pragma unroll
    for (int ni = 0; ni < 2; ++ni)
#pragma unroll
      for (int r = 0; r < 8; ++r) {
        int mg = m0 + wm + mi * 16 + r + hh * 8;
        int ng = n0 + wn + ni * 16 + l;
        float val = acc[mi][ni][r] + bias[ng];
        int b = mg >> 11;
        int t = mg & (SEQ - 1);
        int head = ng >> 6;
        int dh = ng & (DHEAD - 1);
        size_t bh = (size_t)b * NHEAD + head;
        if (which == 2)
          Vt[(bh * DHEAD + dh) * SEQ + t] = (_Float16)val;
        else {
          _Float16* O = (which == 0) ? Qh : Kh;
          O[(bh * SEQ + t) * DHEAD + dh] = (_Float16)val;
        }
      }
}

// ---------------------------------------------------------------------------
// Kernel 2: masked attention, flash style.  One wave = 16 query rows; the
// block's 8 waves share (b,h), so K/V 32-wide tiles are staged once per block
// into double-buffered LDS via async global->LDS DMA, overlapped with compute.
// ---------------------------------------------------------------------------
__global__ __launch_bounds__(256) void attn(
    const _Float16* __restrict__ Qh, const _Float16* __restrict__ Kh,
    const _Float16* __restrict__ Vt, const unsigned char* __restrict__ mask,
    _Float16* __restrict__ Att) {
  __shared__ _Float16 Ks[2][32 * 64];   // [kv][dh]
  __shared__ _Float16 Vs[2][64 * 32];   // [dh][kv]
  __shared__ _Float16 Ps[8 * 16 * 40];  // per-wave 16x32 P tile (stride 40)

  const int tid = threadIdx.x;
  const int lane = tid & 31;
  const int wave = tid >> 5;
  const int l = lane & 15;
  const int hh = lane >> 4;
  const int bh = blockIdx.y;
  const int q0 = blockIdx.x * 128 + wave * 16;
  const size_t base = (size_t)bh * SEQ * DHEAD;
  const _Float16* Vbase = Vt + (size_t)bh * DHEAD * SEQ;

  // Q A-fragments held across the whole KV sweep.
  const _Float16* qrow = Qh + base + (size_t)(q0 + l) * DHEAD;
  const v16h qa0 = frag16(qrow + hh * 8, qrow + 16 + hh * 8);
  const v16h qa1 = frag16(qrow + 32 + hh * 8, qrow + 48 + hh * 8);

  v8f o[4];
#pragma unroll
  for (int f = 0; f < 4; ++f) o[f] = vzero8();
  float mr[8], lr[8];
#pragma unroll
  for (int r = 0; r < 8; ++r) { mr[r] = -1e30f; lr[r] = 0.0f; }

  _Float16* P = Ps + wave * 640;
  const float scale = 0.03125f;  // 1/sqrt(D)
  const float L2E = 1.44269504f;

  const int kr = tid >> 3, kc = (tid & 7) * 8;  // K tile: 32 rows x 8 chunks
  const int vr = tid >> 2, vc = (tid & 3) * 8;  // V tile: 64 rows x 4 chunks

  auto issue_kv = [&](int tile, int buf) {
    const _Float16* gK = Kh + base + (size_t)tile * 32 * DHEAD;
    const _Float16* gV = Vbase + tile * 32;
    async_b128(gK + (size_t)kr * DHEAD + kc, &Ks[buf][kr * 64 + kc]);
    async_b128(gV + (size_t)vr * SEQ + vc, &Vs[buf][vr * 32 + vc]);
  };

  const int nIter = SEQ / 32;
  issue_kv(0, 0);

  for (int it = 0; it < nIter; ++it) {
    const int cur = it & 1;
    const int kv0 = it * 32;

    wait_async0();        // this buffer's tile landed (per wave)
    __syncthreads();      // all waves' tiles landed; prev readers of buf^1 done
    if (it + 1 < nIter) issue_kv(it + 1, cur ^ 1);  // overlap next fill

    v8f s[2];
#pragma unroll
    for (int ns = 0; ns < 2; ++ns) {
      const _Float16* krow = &Ks[cur][(ns * 16 + l) * 64];
      v16h kb0 = frag16(krow + hh * 8, krow + 16 + hh * 8);
      v16h kb1 = frag16(krow + 32 + hh * 8, krow + 48 + hh * 8);
      v8f sv = vzero8();
      sv = wmma16(qa0, kb0, sv);
      sv = wmma16(qa1, kb1, sv);
#pragma unroll
      for (int r = 0; r < 8; ++r) {
        int qi = q0 + r + hh * 8;
        int col = kv0 + ns * 16 + l;
        float x = sv[r] * scale;
        unsigned char mb = mask[(size_t)qi * SEQ + col];
        sv[r] = mb ? -1e30f : x;
      }
      s[ns] = sv;
    }

    float alpha[8];
#pragma unroll
    for (int r = 0; r < 8; ++r) {
      float tm = redmax16(fmaxf(s[0][r], s[1][r]));
      float mn = fmaxf(mr[r], tm);
      alpha[r] = exp2f((mr[r] - mn) * L2E);
      float p0 = exp2f((s[0][r] - mn) * L2E);
      float p1 = exp2f((s[1][r] - mn) * L2E);
      lr[r] = lr[r] * alpha[r] + redsum16(p0 + p1);
      mr[r] = mn;
      P[(r + hh * 8) * 40 + l] = (_Float16)p0;
      P[(r + hh * 8) * 40 + 16 + l] = (_Float16)p1;
    }
#pragma unroll
    for (int f = 0; f < 4; ++f)
#pragma unroll
      for (int r = 0; r < 8; ++r) o[f][r] *= alpha[r];

    // Cross-lane LDS RAW on the wave-private P slab: DS is in-order per wave;
    // drain DScnt and fence the compiler before re-reading P as an A-frag.
    asm volatile("s_wait_dscnt 0x0" ::: "memory");
    const _Float16* prow = P + l * 40;
    v16h pa = frag16(prow + hh * 8, prow + 16 + hh * 8);
    asm volatile("" ::: "memory");

#pragma unroll
    for (int f = 0; f < 4; ++f) {
      const _Float16* vrow = &Vs[cur][(f * 16 + l) * 32];
      v16h vb = frag16(vrow + hh * 8, vrow + 16 + hh * 8);
      o[f] = wmma16(pa, vb, o[f]);
    }
  }

  const int b = bh >> 4;
  const int head = bh & (NHEAD - 1);
#pragma unroll
  for (int f = 0; f < 4; ++f)
#pragma unroll
    for (int r = 0; r < 8; ++r) {
      int qg = q0 + r + hh * 8;
      float val = o[f][r] / lr[r];
      Att[((size_t)b * SEQ + qg) * DMODEL + head * DHEAD + f * 16 + l] =
          (_Float16)val;
    }
}

// ---------------------------------------------------------------------------
// Kernel 3: output projection.  out[8192,1024] f32 = Att(f16) @ Wo^T + bo.
// A tile (already f16) is staged via async global->LDS DMA.
// ---------------------------------------------------------------------------
__global__ __launch_bounds__(256) void out_gemm(
    const _Float16* __restrict__ A, const float* __restrict__ W,
    const float* __restrict__ bias, float* __restrict__ out) {
  __shared__ _Float16 Xs[128 * 40];
  __shared__ _Float16 Ws[128 * 40];

  const int tid = threadIdx.x;
  const int lane = tid & 31;
  const int wave = tid >> 5;
  const int l = lane & 15;
  const int hh = lane >> 4;
  const int m0 = blockIdx.x * 128;
  const int n0 = blockIdx.y * 128;
  const int wm = (wave & 1) * 64;
  const int wn = (wave >> 1) * 32;

  v8f acc[4][2];
#pragma unroll
  for (int mi = 0; mi < 4; ++mi)
#pragma unroll
    for (int ni = 0; ni < 2; ++ni) acc[mi][ni] = vzero8();

  for (int k0 = 0; k0 < DMODEL; k0 += 32) {
#pragma unroll
    for (int i = 0; i < 2; ++i) {  // 128 rows x 4 chunks of 8 halves, async DMA
      int idx = tid + i * 256;
      int row = idx >> 2;
      int c8 = (idx & 3) * 8;
      async_b128(A + (size_t)(m0 + row) * DMODEL + k0 + c8, Xs + row * 40 + c8);
    }
#pragma unroll
    for (int i = 0; i < 4; ++i) {  // W: f32 -> f16 via VGPRs
      int idx = tid + i * 256;
      int row = idx >> 3;
      int c4 = (idx & 7) * 4;
      v4f w4 = *(const v4f*)(W + (size_t)(n0 + row) * DMODEL + k0 + c4);
      _Float16* dw = Ws + row * 40 + c4;
#pragma unroll
      for (int j = 0; j < 4; ++j) dw[j] = (_Float16)w4[j];
    }
    wait_async0();
    __syncthreads();

    v16h af[4], bf[2];
#pragma unroll
    for (int mi = 0; mi < 4; ++mi) {
      const _Float16* ar = Xs + (wm + mi * 16 + l) * 40;
      af[mi] = frag16(ar + hh * 8, ar + 16 + hh * 8);
    }
#pragma unroll
    for (int ni = 0; ni < 2; ++ni) {
      const _Float16* br = Ws + (wn + ni * 16 + l) * 40;
      bf[ni] = frag16(br + hh * 8, br + 16 + hh * 8);
    }
#pragma unroll
    for (int mi = 0; mi < 4; ++mi)
#pragma unroll
      for (int ni = 0; ni < 2; ++ni)
        acc[mi][ni] = wmma16(af[mi], bf[ni], acc[mi][ni]);
    __syncthreads();
  }

#pragma unroll
  for (int mi = 0; mi < 4; ++mi)
#pragma unroll
    for (int ni = 0; ni < 2; ++ni)
#pragma unroll
      for (int r = 0; r < 8; ++r) {
        int mg = m0 + wm + mi * 16 + r + hh * 8;
        int ng = n0 + wn + ni * 16 + l;
        out[(size_t)mg * DMODEL + ng] = acc[mi][ni][r] + bias[ng];
      }
}

extern "C" void kernel_launch(void* const* d_in, const int* in_sizes, int n_in,
                              void* d_out, int out_size, void* d_ws,
                              size_t ws_size, hipStream_t stream) {
  const float* q = (const float*)d_in[0];
  const float* k = (const float*)d_in[1];
  const float* v = (const float*)d_in[2];
  const unsigned char* mask = (const unsigned char*)d_in[3];
  const float* Wq = (const float*)d_in[4];
  const float* bq = (const float*)d_in[5];
  const float* Wk = (const float*)d_in[6];
  const float* bk = (const float*)d_in[7];
  const float* Wv = (const float*)d_in[8];
  const float* bv = (const float*)d_in[9];
  const float* Wo = (const float*)d_in[10];
  const float* bo = (const float*)d_in[11];

  _Float16* ws = (_Float16*)d_ws;
  _Float16* Qh = ws;
  _Float16* Kh = ws + ELEMS;
  _Float16* Vt = ws + 2 * ELEMS;
  _Float16* At = ws + 3 * ELEMS;  // 64 MB total f16 workspace

  dim3 g1(NROWS / 128, DMODEL / 128, 3);
  qkv_gemm<<<g1, 256, 0, stream>>>(q, k, v, Wq, Wk, Wv, bq, bk, bv, Qh, Kh, Vt);

  dim3 g2(SEQ / 128, BATCH * NHEAD, 1);
  attn<<<g2, 256, 0, stream>>>(Qh, Kh, Vt, mask, At);

  dim3 g3(NROWS / 128, DMODEL / 128, 1);
  out_gemm<<<g3, 256, 0, stream>>>(At, Wo, bo, (float*)d_out);
}
